// Proposed_64467459113088
// MI455X (gfx1250) — matrix operand order repaired and checked by reading, not compiled, not run on hardware
//
#include <hip/hip_runtime.h>
#include <math.h>

#define DD 128            // embedding dim D
typedef __attribute__((ext_vector_type(16))) _Float16 v16h;
typedef __attribute__((ext_vector_type(8)))  _Float16 v8h;
typedef __attribute__((ext_vector_type(8)))  float    v8f;

// ---------------------------------------------------------------- utility
__global__ void k_zero(float* p, long n) {
    long i = (long)blockIdx.x * blockDim.x + threadIdx.x;
    if (i < n) p[i] = 0.0f;
}

__global__ void k_init_prop(const float* __restrict__ ue, const float* __restrict__ ve,
                            float* __restrict__ cur, float* __restrict__ acc,
                            int nU, long total) {
    long i = (long)blockIdx.x * blockDim.x + threadIdx.x;
    if (i >= total) return;
    long node = i >> 7;           // /128
    int  d    = (int)(i & 127);
    float v = (node < nU) ? ue[node * DD + d] : ve[(node - nU) * DD + d];
    cur[i] = v;
    acc[i] = v;
}

// COO SpMM: nxt[rows[e]] += vals[e] * cur[cols[e]]   (one wave of 32 lanes per edge,
// each lane owns a float4 chunk of the 128-wide row)
__global__ void k_spmm(const int* __restrict__ rows, const int* __restrict__ cols,
                       const float* __restrict__ vals, const float* __restrict__ cur,
                       float* nxt, int nE) {
    long t = (long)blockIdx.x * blockDim.x + threadIdx.x;
    int e    = (int)(t >> 5);
    int lane = (int)(t & 31);
    if (e >= nE) return;
    int   r = rows[e];
    int   c = cols[e];
    float v = vals[e];
    float4 f = *(const float4*)(cur + (long)c * DD + lane * 4);
    float* dst = nxt + (long)r * DD + lane * 4;
    atomicAdd(dst + 0, v * f.x);
    atomicAdd(dst + 1, v * f.y);
    atomicAdd(dst + 2, v * f.z);
    atomicAdd(dst + 3, v * f.w);
}

__global__ void k_acc_add(float* acc, const float* __restrict__ x, long n) {
    long i = (long)blockIdx.x * blockDim.x + threadIdx.x;
    if (i < n) acc[i] += x[i];
}

__global__ void k_scale(float* p, float s, long n) {
    long i = (long)blockIdx.x * blockDim.x + threadIdx.x;
    if (i < n) p[i] *= s;
}

// ---------------------------------------------------------------- WMMA GEMM
// Out[m,n] = epilogue( sum_k A[ar(m), k] * W[k,n] )
//   ar(m)  = GID ? gidx[m] : m
//   CIN: 0 = none; 1 = add Cin before activation; 2 = add Cin after activation
//   Cin row = CGID ? cgidx[m] : m
// Block = 2*N threads (N/16 waves), computes a 32 x N tile; K stepped by 32.
// A tile (32x32) and B slab (32xN, stored transposed Bs[n][k]) staged in LDS as f16;
// fragments read as aligned b128 LDS loads; 2 x v_wmma_f32_16x16x32_f16 per wave/K-step.
template<int N_T, bool GID, int CIN, bool CGID, bool ACT>
__global__ __launch_bounds__(2 * N_T)
void k_gemm_wmma(const float* __restrict__ A, const int* __restrict__ gidx, int lda,
                 const float* __restrict__ W, const float* __restrict__ bias,
                 const float* Cin, const int* __restrict__ cgidx,
                 float* Out, int K) {
    constexpr int BT = 2 * N_T;
    const int tid  = threadIdx.x;
    const int lane = tid & 31;
    const int wave = tid >> 5;            // one 16-wide N slice per wave
    const int m0   = blockIdx.x * 32;
    const int n0   = wave << 4;
    const int l16  = lane & 15;
    const int h    = lane >> 4;
    const int bn   = n0 + l16;

    __shared__ __align__(64) _Float16 As[32 * 32];
    __shared__ __align__(64) _Float16 Bs[N_T * 32];
    __shared__ int arows[32];
    __shared__ int crows[32];

    if (tid < 32) {
        int m = m0 + tid;
        arows[tid] = GID ? gidx[m] : m;
        if (CIN != 0) crows[tid] = CGID ? cgidx[m] : m;
    }
    __syncthreads();

    v8f c0 = {};
    v8f c1 = {};
    for (int k0 = 0; k0 < K; k0 += 32) {
        // ---- stage A tile (32 rows x 32 k), f32 -> f16, coalesced float2 per thread
#pragma unroll
        for (int p = tid; p < 512; p += BT) {
            const int row = p >> 4;
            const int cp  = (p & 15) << 1;
            const float2 f = *(const float2*)(A + (long)arows[row] * lda + k0 + cp);
            As[row * 32 + cp]     = (_Float16)f.x;
            As[row * 32 + cp + 1] = (_Float16)f.y;
        }
        // ---- stage B slab transposed: Bs[n][k - k0]; rows of W are lane-coalesced
        {
            const int n  = tid & (N_T - 1);
            const int kh = tid / N_T;     // 0 or 1
            const float* pw = W + (long)(k0 + 16 * kh) * N_T + n;
            if (k0 + 32 < K) __builtin_prefetch(pw + (long)32 * N_T, 0, 0);
            _Float16* dst = &Bs[n * 32 + 16 * kh];
#pragma unroll
            for (int i = 0; i < 16; ++i) dst[i] = (_Float16)pw[(long)i * N_T];
        }
        __syncthreads();
        // ---- fragments (contiguous k = 16h + s in A and B) + 2 WMMAs
        const _Float16* ap0 = &As[l16 * 32 + 16 * h];
        const _Float16* ap1 = &As[(16 + l16) * 32 + 16 * h];
        v8h a0lo = *(const v8h*)(ap0);
        v8h a0hi = *(const v8h*)(ap0 + 8);
        v8h a1lo = *(const v8h*)(ap1);
        v8h a1hi = *(const v8h*)(ap1 + 8);
        v16h a0 = __builtin_shufflevector(a0lo, a0hi, 0,1,2,3,4,5,6,7,8,9,10,11,12,13,14,15);
        v16h a1 = __builtin_shufflevector(a1lo, a1hi, 0,1,2,3,4,5,6,7,8,9,10,11,12,13,14,15);
        v16h b  = *(const v16h*)(&Bs[bn * 32 + 16 * h]);
        c0 = __builtin_amdgcn_wmma_f32_16x16x32_f16(false, a0, false, b, (short)0, c0, false, false);
        c1 = __builtin_amdgcn_wmma_f32_16x16x32_f16(false, a1, false, b, (short)0, c1, false, false);
        __syncthreads();
    }
    const float bv = bias ? bias[bn] : 0.0f;
#pragma unroll
    for (int j = 0; j < 8; ++j) {
        const int mm = 8 * h + j;         // C/D layout: VGPR j -> M = 8h + j, N = lane&15
        {
            float v = c0[j] + bv;
            if (CIN == 1) v += Cin[(long)crows[mm] * N_T + bn];
            if (ACT) v = fmaxf(v, 0.0f);
            if (CIN == 2) v += Cin[(long)crows[mm] * N_T + bn];
            Out[(long)(m0 + mm) * N_T + bn] = v;
        }
        {
            const int mm1 = mm + 16;
            float v = c1[j] + bv;
            if (CIN == 1) v += Cin[(long)crows[mm1] * N_T + bn];
            if (ACT) v = fmaxf(v, 0.0f);
            if (CIN == 2) v += Cin[(long)crows[mm1] * N_T + bn];
            Out[(long)(m0 + mm1) * N_T + bn] = v;
        }
    }
}

// ---------------------------------------------------------------- BatchNorm(train) + ReLU
__global__ void k_bn_relu(const float* __restrict__ x, const float* __restrict__ g,
                          const float* __restrict__ b, float* __restrict__ y,
                          int M, int N) {
    const int j = blockIdx.x;
    __shared__ float s1[256];
    __shared__ float s2[256];
    float sum = 0.0f, sq = 0.0f;
    for (int r = threadIdx.x; r < M; r += blockDim.x) {
        float v = x[(long)r * N + j];
        sum += v; sq += v * v;
    }
    s1[threadIdx.x] = sum; s2[threadIdx.x] = sq;
    __syncthreads();
    for (int s = 128; s > 0; s >>= 1) {
        if (threadIdx.x < s) { s1[threadIdx.x] += s1[threadIdx.x + s]; s2[threadIdx.x] += s2[threadIdx.x + s]; }
        __syncthreads();
    }
    const float mean = s1[0] / (float)M;
    const float var  = s2[0] / (float)M - mean * mean;
    const float scale = g[j] * rsqrtf(var + 1e-5f);
    const float shift = b[j] - mean * scale;
    for (int r = threadIdx.x; r < M; r += blockDim.x) {
        float v = x[(long)r * N + j] * scale + shift;
        y[(long)r * N + j] = fmaxf(v, 0.0f);
    }
}

// ---------------------------------------------------------------- final 64->1 + sigmoid
__global__ void k_final(const float* __restrict__ x, const float* __restrict__ W,
                        const float* __restrict__ b, float* out, int M, int Kc, int col) {
    int m = blockIdx.x * blockDim.x + threadIdx.x;
    if (m >= M) return;
    float s = b[0];
    const float* xr = x + (long)m * Kc;
#pragma unroll 8
    for (int k = 0; k < Kc; ++k) s += xr[k] * W[k];
    out[(long)m * 2 + col] = 1.0f / (1.0f + expf(-s));
}

// ---------------------------------------------------------------- host
extern "C" void kernel_launch(void* const* d_in, const int* in_sizes, int n_in,
                              void* d_out, int out_size, void* d_ws, size_t ws_size,
                              hipStream_t stream) {
    const float* su_emb = (const float*)d_in[0];
    const float* tu_emb = (const float*)d_in[1];
    const float* sv_emb = (const float*)d_in[2];
    const float* tv_emb = (const float*)d_in[3];
    const float* s_text = (const float*)d_in[4];
    const float* s_vis  = (const float*)d_in[5];
    const float* s_rev  = (const float*)d_in[6];
    const float* t_text = (const float*)d_in[7];
    const float* t_vis  = (const float*)d_in[8];
    const float* t_rev  = (const float*)d_in[9];
    const float* Wst = (const float*)d_in[10]; const float* bst = (const float*)d_in[11];
    const float* Wsv = (const float*)d_in[12]; const float* bsv = (const float*)d_in[13];
    const float* Wsr = (const float*)d_in[14]; const float* bsr = (const float*)d_in[15];
    const float* Wtt = (const float*)d_in[16]; const float* btt = (const float*)d_in[17];
    const float* Wtv = (const float*)d_in[18]; const float* btv = (const float*)d_in[19];
    const float* Wtr = (const float*)d_in[20]; const float* btr = (const float*)d_in[21];
    const float* Wsc = (const float*)d_in[22]; const float* bsc = (const float*)d_in[23];
    const float* Wtc = (const float*)d_in[24]; const float* btc = (const float*)d_in[25];
    const float* Wsh = (const float*)d_in[26]; const float* bsh = (const float*)d_in[27];
    const float* Wps = (const float*)d_in[28]; const float* bps = (const float*)d_in[29];
    const float* Wpt = (const float*)d_in[30]; const float* bpt = (const float*)d_in[31];
    const float* Wsp1 = (const float*)d_in[32]; const float* bsp1 = (const float*)d_in[33];
    const float* gs1  = (const float*)d_in[34]; const float* hs1  = (const float*)d_in[35];
    const float* Wsp2 = (const float*)d_in[36]; const float* bsp2 = (const float*)d_in[37];
    const float* gs2  = (const float*)d_in[38]; const float* hs2  = (const float*)d_in[39];
    const float* Wsp3 = (const float*)d_in[40]; const float* bsp3 = (const float*)d_in[41];
    const float* Wtp1 = (const float*)d_in[42]; const float* btp1 = (const float*)d_in[43];
    const float* gt1  = (const float*)d_in[44]; const float* ht1  = (const float*)d_in[45];
    const float* Wtp2 = (const float*)d_in[46]; const float* btp2 = (const float*)d_in[47];
    const float* gt2  = (const float*)d_in[48]; const float* ht2  = (const float*)d_in[49];
    const float* Wtp3 = (const float*)d_in[50]; const float* btp3 = (const float*)d_in[51];
    const float* s_vals = (const float*)d_in[52];
    const float* t_vals = (const float*)d_in[53];
    const int* s_rows = (const int*)d_in[54];
    const int* s_cols = (const int*)d_in[55];
    const int* t_rows = (const int*)d_in[56];
    const int* t_cols = (const int*)d_in[57];
    const int* users   = (const int*)d_in[58];
    const int* s_items = (const int*)d_in[59];
    const int* t_items = (const int*)d_in[60];

    const int U_  = in_sizes[0] / DD;   // 50000
    const int IS_ = in_sizes[2] / DD;   // 30000
    const int IT_ = in_sizes[3] / DD;   // 30000
    const int Bn  = in_sizes[58];       // 8192
    const int nEs = in_sizes[54];       // 2E
    const int nEt = in_sizes[56];
    const int DT_ = 384, DV_ = 512, DR_ = 768;
    const long nodesS = (long)U_ + IS_;
    const long nodesT = (long)U_ + IT_;
    const long maxNodes = nodesS > nodesT ? nodesS : nodesT;

    // ---- carve workspace
    float* wsp = (float*)d_ws;
    size_t off = 0;
    auto carve = [&](long nf) { float* p = wsp + off; off += (size_t)((nf + 63) & ~63L); return p; };
    float* acc_s = carve(nodesS * DD);
    float* acc_t = carve(nodesT * DD);
    float* cur   = carve(maxNodes * DD);
    float* nxt   = carve(maxNodes * DD);
    float* tmpf  = carve((long)Bn * DD);
    float* s_feat = carve((long)Bn * DD);
    float* t_feat = carve((long)Bn * DD);
    float* sv_f   = carve((long)Bn * DD);
    float* tv_f   = carve((long)Bn * DD);
    float* su_f   = carve((long)Bn * DD);
    float* tu_f   = carve((long)Bn * DD);
    float* sh_b   = carve((long)Bn * DD);
    float* s_user = carve((long)Bn * DD);
    float* t_user = carve((long)Bn * DD);
    float* x1  = carve((long)Bn * DD);
    float* x1n = carve((long)Bn * DD);
    float* x2  = carve((long)Bn * 64);
    float* x2n = carve((long)Bn * 64);
    float* outp = (float*)d_out;

    // ---- LightGCN propagation (bandwidth/atomic bound)
    auto propagate = [&](const float* ue, const float* ve, const int* rows, const int* cols,
                         const float* vals, int nU, long nNodes, int nE, float* acc) {
        long total = nNodes * DD;
        unsigned blk = (unsigned)((total + 255) / 256);
        k_init_prop<<<blk, 256, 0, stream>>>(ue, ve, cur, acc, nU, total);
        float* a = cur; float* b2 = nxt;
        long tE = (long)nE * 32;
        unsigned eblk = (unsigned)((tE + 255) / 256);
        for (int l = 0; l < 3; ++l) {
            k_zero<<<blk, 256, 0, stream>>>(b2, total);
            k_spmm<<<eblk, 256, 0, stream>>>(rows, cols, vals, a, b2, nE);
            k_acc_add<<<blk, 256, 0, stream>>>(acc, b2, total);
            float* t2 = a; a = b2; b2 = t2;
        }
        k_scale<<<blk, 256, 0, stream>>>(acc, 0.25f, total);
    };
    propagate(su_emb, sv_emb, s_rows, s_cols, s_vals, U_, nodesS, nEs, acc_s);
    propagate(tu_emb, tv_emb, t_rows, t_cols, t_vals, U_, nodesT, nEt, acc_t);

    // ---- dense pipeline (templated WMMA GEMM, 32xN tile per block)
    const unsigned g32 = (unsigned)(Bn / 32);

    // multimodal item features: relu(text@Wt+bt) + relu(vis@Wv+bv)
    k_gemm_wmma<128, true, 0, false, true ><<<g32, 256, 0, stream>>>(s_text, s_items, DT_, Wst, bst, nullptr, nullptr, tmpf,   DT_);
    k_gemm_wmma<128, true, 2, false, true ><<<g32, 256, 0, stream>>>(s_vis,  s_items, DV_, Wsv, bsv, tmpf,    nullptr, s_feat, DV_);
    k_gemm_wmma<128, true, 0, false, true ><<<g32, 256, 0, stream>>>(t_text, t_items, DT_, Wtt, btt, nullptr, nullptr, tmpf,   DT_);
    k_gemm_wmma<128, true, 2, false, true ><<<g32, 256, 0, stream>>>(t_vis,  t_items, DV_, Wtv, btv, tmpf,    nullptr, t_feat, DV_);
    // item fuse: concat([v_graph, feat]) @ Wc + bc  (K split into two accumulating calls)
    k_gemm_wmma<128, true, 0, false, false><<<g32, 256, 0, stream>>>(acc_s + (long)U_ * DD, s_items, DD, Wsc,          bsc,     nullptr, nullptr, sv_f, DD);
    k_gemm_wmma<128, false, 1, false, false><<<g32, 256, 0, stream>>>(s_feat, nullptr, DD,          Wsc + DD * DD, nullptr, sv_f,    nullptr, sv_f, DD);
    k_gemm_wmma<128, true, 0, false, false><<<g32, 256, 0, stream>>>(acc_t + (long)U_ * DD, t_items, DD, Wtc,          btc,     nullptr, nullptr, tv_f, DD);
    k_gemm_wmma<128, false, 1, false, false><<<g32, 256, 0, stream>>>(t_feat, nullptr, DD,          Wtc + DD * DD, nullptr, tv_f,    nullptr, tv_f, DD);
    // user enrich: u_graph[users] + relu(rev[users]@Wr+br)
    k_gemm_wmma<128, true, 2, true, true ><<<g32, 256, 0, stream>>>(s_rev, users, DR_, Wsr, bsr, acc_s, users, su_f, DR_);
    k_gemm_wmma<128, true, 2, true, true ><<<g32, 256, 0, stream>>>(t_rev, users, DR_, Wtr, btr, acc_t, users, tu_f, DR_);
    // shared = relu(concat([su_f,tu_f])@Wsh+bsh)
    k_gemm_wmma<128, false, 0, false, false><<<g32, 256, 0, stream>>>(su_f, nullptr, DD, Wsh,          bsh,     nullptr, nullptr, sh_b, DD);
    k_gemm_wmma<128, false, 1, false, true ><<<g32, 256, 0, stream>>>(tu_f, nullptr, DD, Wsh + DD * DD, nullptr, sh_b,   nullptr, sh_b, DD);
    // user reps = shared + relu(u_f@Wp+bp)
    k_gemm_wmma<128, false, 2, false, true ><<<g32, 256, 0, stream>>>(su_f, nullptr, DD, Wps, bps, sh_b, nullptr, s_user, DD);
    k_gemm_wmma<128, false, 2, false, true ><<<g32, 256, 0, stream>>>(tu_f, nullptr, DD, Wpt, bpt, sh_b, nullptr, t_user, DD);

    // predictor S
    k_gemm_wmma<128, false, 0, false, false><<<g32, 256, 0, stream>>>(s_user, nullptr, DD, Wsp1,          bsp1,    nullptr, nullptr, x1, DD);
    k_gemm_wmma<128, false, 1, false, false><<<g32, 256, 0, stream>>>(sv_f,   nullptr, DD, Wsp1 + DD * DD, nullptr, x1,     nullptr, x1, DD);
    k_bn_relu<<<DD, 256, 0, stream>>>(x1, gs1, hs1, x1n, Bn, DD);
    k_gemm_wmma<64, false, 0, false, false><<<g32, 128, 0, stream>>>(x1n, nullptr, DD, Wsp2, bsp2, nullptr, nullptr, x2, DD);
    k_bn_relu<<<64, 256, 0, stream>>>(x2, gs2, hs2, x2n, Bn, 64);
    k_final<<<(Bn + 255) / 256, 256, 0, stream>>>(x2n, Wsp3, bsp3, outp, Bn, 64, 0);

    // predictor T
    k_gemm_wmma<128, false, 0, false, false><<<g32, 256, 0, stream>>>(t_user, nullptr, DD, Wtp1,          btp1,    nullptr, nullptr, x1, DD);
    k_gemm_wmma<128, false, 1, false, false><<<g32, 256, 0, stream>>>(tv_f,   nullptr, DD, Wtp1 + DD * DD, nullptr, x1,     nullptr, x1, DD);
    k_bn_relu<<<DD, 256, 0, stream>>>(x1, gt1, ht1, x1n, Bn, DD);
    k_gemm_wmma<64, false, 0, false, false><<<g32, 128, 0, stream>>>(x1n, nullptr, DD, Wtp2, btp2, nullptr, nullptr, x2, DD);
    k_bn_relu<<<64, 256, 0, stream>>>(x2, gt2, ht2, x2n, Bn, 64);
    k_final<<<(Bn + 255) / 256, 256, 0, stream>>>(x2n, Wtp3, btp3, outp, Bn, 64, 1);
}
